// GRU_78039555768457
// MI455X (gfx1250) — compile-verified, matrix operand order checked by reading, not run on hardware
//
#include <hip/hip_runtime.h>

// Fused GRU cell for MI455X (gfx1250, wave32, WMMA).
// Compute-bound (≈1000 FLOP/byte): use bf16x3 split-precision WMMA
// (hi*hi + lo*hi + hi*lo) on V_WMMA_F32_16X16X32_BF16 for ~fp32 accuracy
// at 8x the per-instruction MACs of the fp32 WMMA path.

#define DIM   1024
#define NROWS 32768
#define BM    128
#define BN    64
#define BK    32

typedef __attribute__((ext_vector_type(16))) __bf16         v16bf;
typedef __attribute__((ext_vector_type(8)))  float          v8f;
typedef __attribute__((ext_vector_type(4)))  float          f32x4;
typedef __attribute__((ext_vector_type(4)))  unsigned int   u32x4;
typedef __attribute__((ext_vector_type(4)))  unsigned short u16x4;

__device__ __forceinline__ v8f mm_bf16(v16bf a, v16bf b, v8f c) {
  // (neg_a, A, neg_b, B, c_mod, C, reuse_a, reuse_b)
  return __builtin_amdgcn_wmma_f32_16x16x32_bf16(false, a, false, b, (short)0, c,
                                                 false, false);
}

// Fragment gather matching the 16-bit A/B VGPR layout:
// lane L -> matrix row (L & 15), K groups {k0..k0+7, k0+16..k0+23}, k0=(L>>4)*8.
// Two 16-byte LDS reads per fragment (ds_load_b128).
__device__ __forceinline__ v16bf ld_frag(const unsigned short* s, int row0,
                                         int lm, int lk) {
  const unsigned short* p = s + (row0 + lm) * BK;
  union { v16bf v; u32x4 q[2]; } u;
  u.q[0] = *(const u32x4*)(p + lk);
  u.q[1] = *(const u32x4*)(p + lk + 16);
  return u.v;
}

// fp32 -> (hi, lo) bf16 split: hi = truncate(x), lo = bf16(x - hi).
__device__ __forceinline__ void split_store4(f32x4 v, unsigned short* hi,
                                             unsigned short* lo) {
  u16x4 h, l;
#pragma unroll
  for (int j = 0; j < 4; ++j) {
    unsigned bits = __float_as_uint(v[j]);
    h[j] = (unsigned short)(bits >> 16);
    float rem = v[j] - __uint_as_float(bits & 0xffff0000u);
    l[j] = (unsigned short)(__float_as_uint(rem) >> 16);
  }
  *(u16x4*)hi = h;
  *(u16x4*)lo = l;
}

__device__ __forceinline__ float sigmoidf_(float x) {
  return 1.0f / (1.0f + __expf(-x));
}

__global__ void __launch_bounds__(256)
gru_fused_wmma(const float* __restrict__ X, const float* __restrict__ H,
               const float* __restrict__ Wir, const float* __restrict__ Whr,
               const float* __restrict__ Wiz, const float* __restrict__ Whz,
               const float* __restrict__ Win, const float* __restrict__ Whn,
               const float* __restrict__ bir, const float* __restrict__ bhr,
               const float* __restrict__ biz, const float* __restrict__ bhz,
               const float* __restrict__ bin_, const float* __restrict__ bhn,
               float* __restrict__ Out) {
  extern __shared__ unsigned short smem[];
  unsigned short* sXhi = smem;            // 128x32 bf16
  unsigned short* sXlo = smem + 4096;
  unsigned short* sHhi = smem + 8192;
  unsigned short* sHlo = smem + 12288;
  unsigned short* sW   = smem + 16384;    // 12 tiles of 64x32 (hi[0..5], lo[0..5])

  const int tid  = threadIdx.x;
  const int lane = tid & 31;
  const int wave = tid >> 5;
  const int lm = lane & 15;
  const int lk = (lane >> 4) * 8;

  const int rowBase = blockIdx.y * BM;
  const int colBase = blockIdx.x * BN;
  const int rowOff  = (wave & 3) * 32;    // 4 waves over M
  const int colOff  = (wave >> 2) * 32;   // 2 waves over N

  const float* const Wm[6] = {Wir, Whr, Wiz, Whz, Win, Whn};

  // acc[mi][ni][gate]: gate 0 = r (x*Wir + h*Whr), 1 = z, 2 = x*Win, 3 = h*Whn
  v8f acc[2][2][4];
  const v8f vzero = {0.f, 0.f, 0.f, 0.f, 0.f, 0.f, 0.f, 0.f};
#pragma unroll
  for (int mi = 0; mi < 2; ++mi)
#pragma unroll
    for (int ni = 0; ni < 2; ++ni)
#pragma unroll
      for (int g = 0; g < 4; ++g) acc[mi][ni][g] = vzero;

  for (int kc = 0; kc < DIM; kc += BK) {
    // ---- stage A tiles (x and h): 128x32 fp32 -> hi/lo bf16 in LDS ----
#pragma unroll
    for (int it = 0; it < (BM * BK) / (4 * 256); ++it) {   // 4 iters
      int i   = tid + it * 256;
      int row = (i * 4) >> 5;
      int col = (i * 4) & 31;
      const float* px = X + (size_t)(rowBase + row) * DIM + kc + col;
      const float* ph = H + (size_t)(rowBase + row) * DIM + kc + col;
      if (kc + BK < DIM) {
        __builtin_prefetch(px + BK, 0, 0);
        __builtin_prefetch(ph + BK, 0, 0);
      }
      split_store4(*(const f32x4*)px, sXhi + row * BK + col, sXlo + row * BK + col);
      split_store4(*(const f32x4*)ph, sHhi + row * BK + col, sHlo + row * BK + col);
    }
    // ---- stage the six 64x32 weight tiles ----
#pragma unroll
    for (int g = 0; g < 6; ++g) {
      const float* Wp = Wm[g];
      unsigned short* dhi = sW + g * (BN * BK);
      unsigned short* dlo = sW + (6 + g) * (BN * BK);
#pragma unroll
      for (int it = 0; it < (BN * BK) / (4 * 256); ++it) { // 2 iters
        int i   = tid + it * 256;
        int row = (i * 4) >> 5;
        int col = (i * 4) & 31;
        const float* pw = Wp + (size_t)(colBase + row) * DIM + kc + col;
        if (kc + BK < DIM) __builtin_prefetch(pw + BK, 0, 0);
        split_store4(*(const f32x4*)pw, dhi + row * BK + col, dlo + row * BK + col);
      }
    }
    __syncthreads();

    // ---- WMMA compute: 72 v_wmma per wave per K-chunk ----
#pragma unroll
    for (int ni = 0; ni < 2; ++ni) {
      const int n0 = colOff + ni * 16;
      v16bf bh[6], bl[6];
#pragma unroll
      for (int g = 0; g < 6; ++g) {
        bh[g] = ld_frag(sW + g * (BN * BK), n0, lm, lk);
        bl[g] = ld_frag(sW + (6 + g) * (BN * BK), n0, lm, lk);
      }
#pragma unroll
      for (int mi = 0; mi < 2; ++mi) {
        const int m0 = rowOff + mi * 16;
        v16bf xh = ld_frag(sXhi, m0, lm, lk);
        v16bf xl = ld_frag(sXlo, m0, lm, lk);
        v16bf hh = ld_frag(sHhi, m0, lm, lk);
        v16bf hl = ld_frag(sHlo, m0, lm, lk);
        v8f* a = acc[mi][ni];
        // r gate: x*Wir + h*Whr, bf16x3 each
        a[0] = mm_bf16(xh, bh[0], a[0]); a[0] = mm_bf16(xl, bh[0], a[0]); a[0] = mm_bf16(xh, bl[0], a[0]);
        a[0] = mm_bf16(hh, bh[1], a[0]); a[0] = mm_bf16(hl, bh[1], a[0]); a[0] = mm_bf16(hh, bl[1], a[0]);
        // z gate: x*Wiz + h*Whz
        a[1] = mm_bf16(xh, bh[2], a[1]); a[1] = mm_bf16(xl, bh[2], a[1]); a[1] = mm_bf16(xh, bl[2], a[1]);
        a[1] = mm_bf16(hh, bh[3], a[1]); a[1] = mm_bf16(hl, bh[3], a[1]); a[1] = mm_bf16(hh, bl[3], a[1]);
        // candidate, input side: x*Win
        a[2] = mm_bf16(xh, bh[4], a[2]); a[2] = mm_bf16(xl, bh[4], a[2]); a[2] = mm_bf16(xh, bl[4], a[2]);
        // candidate, hidden side: h*Whn (kept separate for the r-gate multiply)
        a[3] = mm_bf16(hh, bh[5], a[3]); a[3] = mm_bf16(hl, bh[5], a[3]); a[3] = mm_bf16(hh, bl[5], a[3]);
      }
    }
    __syncthreads();   // protect LDS before next chunk's stores
  }

  // ---- epilogue: biases, gates, output (written twice: tuple (h_t, h_t)) ----
  // C/D layout: VGPR e holds M = e (lanes 0-15) or M = e + 8 (lanes 16-31); N = lane & 15.
#pragma unroll
  for (int ni = 0; ni < 2; ++ni) {
    const int gcol = colBase + colOff + ni * 16 + lm;
    const float vbr = bir[gcol] + bhr[gcol];
    const float vbz = biz[gcol] + bhz[gcol];
    const float vbi = bin_[gcol];
    const float vbh = bhn[gcol];
#pragma unroll
    for (int mi = 0; mi < 2; ++mi) {
      const v8f ar = acc[mi][ni][0];
      const v8f az = acc[mi][ni][1];
      const v8f ai = acc[mi][ni][2];
      const v8f ah = acc[mi][ni][3];
#pragma unroll
      for (int e = 0; e < 8; ++e) {
        const int grow = rowBase + rowOff + mi * 16 + lk + e;
        const float hv = H[(size_t)grow * DIM + gcol];
        const float r  = sigmoidf_(ar[e] + vbr);
        const float z  = sigmoidf_(az[e] + vbz);
        const float n  = tanhf(ai[e] + vbi + r * (ah[e] + vbh));
        const float ht = (1.0f - z) * n + z * hv;
        Out[(size_t)grow * DIM + gcol] = ht;
        Out[(size_t)NROWS * DIM + (size_t)grow * DIM + gcol] = ht;
      }
    }
  }
}

extern "C" void kernel_launch(void* const* d_in, const int* in_sizes, int n_in,
                              void* d_out, int out_size, void* d_ws, size_t ws_size,
                              hipStream_t stream) {
  (void)in_sizes; (void)n_in; (void)out_size; (void)d_ws; (void)ws_size;
  const float* X   = (const float*)d_in[0];
  const float* H   = (const float*)d_in[1];
  const float* Wir = (const float*)d_in[2];
  const float* bir = (const float*)d_in[3];
  const float* Whr = (const float*)d_in[4];
  const float* bhr = (const float*)d_in[5];
  const float* Wiz = (const float*)d_in[6];
  const float* biz = (const float*)d_in[7];
  const float* Whz = (const float*)d_in[8];
  const float* bhz = (const float*)d_in[9];
  const float* Win = (const float*)d_in[10];
  const float* bin_= (const float*)d_in[11];
  const float* Whn = (const float*)d_in[12];
  const float* bhn = (const float*)d_in[13];
  float* Out = (float*)d_out;

  const size_t shbytes = (16384 + 12 * (size_t)(BN * BK)) * sizeof(unsigned short); // 80 KB
  hipFuncSetAttribute((const void*)gru_fused_wmma,
                      hipFuncAttributeMaxDynamicSharedMemorySize, (int)shbytes);

  dim3 grid(DIM / BN, NROWS / BM);   // 16 x 256 workgroups
  gru_fused_wmma<<<grid, 256, shbytes, stream>>>(
      X, H, Wir, Whr, Wiz, Whz, Win, Whn,
      bir, bhr, biz, bhz, bin_, bhn, Out);
}
